// PredictionNetwork_48026324304616
// MI455X (gfx1250) — matrix operand order repaired and checked by reading, not compiled, not run on hardware
//
#include <hip/hip_runtime.h>
#include <hip/hip_bf16.h>

typedef _Float16 v16h __attribute__((ext_vector_type(16)));
typedef _Float16 h8   __attribute__((ext_vector_type(8)));
typedef _Float16 h2   __attribute__((ext_vector_type(2)));
typedef float    v8f  __attribute__((ext_vector_type(8)));

#define CIN   512
#define HID   256
#define HW    4096
#define NB    32
#define NA    9
#define NC    20
#define NOUT  65      // 5*A + C
#define NOUTP 80      // padded to 5 WMMA M-tiles
#define MIDX  65536

// workspace layout (bytes)
#define WS_W1H  0                      // 256*512 f16  = 262144 B
#define WS_W2H  262144                 // 80*256 f16   = 40960 B
#define WS_OUT  303104                 // 131072*65 f32 = 34078720 B

// ---------------------------------------------------------------------------
// One-time weight conversion: W1 -> f16, W2 -> f16 zero-padded to 80 rows.
// ---------------------------------------------------------------------------
__global__ __launch_bounds__(256) void prep_weights(
    const float* __restrict__ W1, const float* __restrict__ W2,
    _Float16* __restrict__ w1h, _Float16* __restrict__ w2h) {
  int t = blockIdx.x * blockDim.x + threadIdx.x;
  if (t < HID * CIN) {
    w1h[t] = (_Float16)W1[t];
  }
  int j = t - HID * CIN;
  if (j >= 0 && j < NOUTP * HID) {
    int row = j / HID;
    int col = j - row * HID;
    w2h[j] = (row < NOUT) ? (_Float16)W2[row * HID + col] : (_Float16)0.0f;
  }
}

// ---------------------------------------------------------------------------
// Fused 2-layer MLP over 64 positions per block. 8 waves (wave32).
// GEMM1: h[256 x 64] = leakyrelu(W1(256x512) @ X(512x64) + b1), f16 in LDS.
// GEMM2: out[65 x 64] = W2p(80x256) @ h + b2, fp32 to workspace (pos-major).
// ---------------------------------------------------------------------------
__global__ __launch_bounds__(256, 1) void fused_mlp(
    const float* __restrict__ X, const _Float16* __restrict__ W1h,
    const float* __restrict__ b1, const _Float16* __restrict__ W2h,
    const float* __restrict__ b2, float* __restrict__ outbuf) {
  __shared__ _Float16 ldsB[64 * 40];   // staged feature tile [s][k], stride 40
  __shared__ _Float16 hbuf[64 * 264];  // hidden activations [s][o], stride 264

  const int wg    = blockIdx.x;        // 0..2047
  const int b     = wg >> 6;           // batch image
  const int sbase = (wg & 63) << 6;    // 64-position tile within image
  const int t     = threadIdx.x;
  const int wave  = t >> 5;
  const int lane  = t & 31;
  const int lr    = lane & 15;
  const int lhi   = lane >> 4;         // 0 or 1
  const int kbA   = lhi * 8;           // A-fragment K chunk base
  const int kbB   = lhi * 8;           // B-fragment K chunk base
  const int Mbase = wave * 32;

  const float* Xb = X + (size_t)b * CIN * HW + sbase;

  v8f acc[2][4];
  {
    v8f z = {};
    #pragma unroll
    for (int mt = 0; mt < 2; ++mt)
      #pragma unroll
      for (int nt = 0; nt < 4; ++nt) acc[mt][nt] = z;
  }

  for (int kt = 0; kt < CIN / 32; ++kt) {
    const int cbase = kt * 32;
    // stage 32(k) x 64(s) fp32 -> f16 transposed into ldsB[s][k]
    {
      const int s  = t & 63;
      const int kq = t >> 6;  // 0..3 -> 8 channels each
      const float* src = Xb + (size_t)(cbase + kq * 8) * HW + s;
      h8 v;
      #pragma unroll
      for (int u = 0; u < 8; ++u) v[u] = (_Float16)src[(size_t)u * HW];
      *(h8*)(&ldsB[s * 40 + kq * 8]) = v;
      if (kt + 1 < CIN / 32) __builtin_prefetch(src + (size_t)32 * HW, 0, 0);
    }
    __syncthreads();

    // A fragments: W1 f16, rows Mbase+mt*16+lr, K layout {kb..kb+7, 16+kb..}
    v16h afrag[2];
    #pragma unroll
    for (int mt = 0; mt < 2; ++mt) {
      const _Float16* ap =
          W1h + (size_t)(Mbase + mt * 16 + lr) * CIN + cbase + kbA;
      h8 lo = *(const h8*)ap;
      h8 hi = *(const h8*)(ap + 16);
      afrag[mt] = __builtin_shufflevector(lo, hi, 0, 1, 2, 3, 4, 5, 6, 7, 8, 9,
                                          10, 11, 12, 13, 14, 15);
    }
    // B fragments from LDS: col = nt*16+lr, K layout {kb..kb+7, 16+kb..}
    #pragma unroll
    for (int nt = 0; nt < 4; ++nt) {
      const _Float16* bp = &ldsB[(nt * 16 + lr) * 40 + kbB];
      h8 lo = *(const h8*)bp;
      h8 hi = *(const h8*)(bp + 16);
      v16h bfrag = __builtin_shufflevector(lo, hi, 0, 1, 2, 3, 4, 5, 6, 7, 8, 9,
                                           10, 11, 12, 13, 14, 15);
      #pragma unroll
      for (int mt = 0; mt < 2; ++mt)
        acc[mt][nt] = __builtin_amdgcn_wmma_f32_16x16x32_f16(
            false, afrag[mt], false, bfrag, (short)0, acc[mt][nt], false,
            false);
    }
    __syncthreads();
  }

  // GEMM1 epilogue: +b1, leaky relu, f16 -> hbuf[s][o]
  #pragma unroll
  for (int mt = 0; mt < 2; ++mt) {
    #pragma unroll
    for (int nt = 0; nt < 4; ++nt) {
      const int s = nt * 16 + lr;
      #pragma unroll
      for (int e = 0; e < 8; e += 2) {
        const int o = Mbase + mt * 16 + lhi * 8 + e;
        float v0 = acc[mt][nt][e] + b1[o];
        float v1 = acc[mt][nt][e + 1] + b1[o + 1];
        v0 = v0 > 0.0f ? v0 : 0.01f * v0;
        v1 = v1 > 0.0f ? v1 : 0.01f * v1;
        h2 p = {(_Float16)v0, (_Float16)v1};
        *(h2*)(&hbuf[s * 264 + o]) = p;
      }
    }
  }
  __syncthreads();

  // GEMM2: waves 0..4 each own one 16-row tile of the 80-row padded output
  if (wave < 5) {
    v8f acc2[4];
    {
      v8f z = {};
      #pragma unroll
      for (int nt = 0; nt < 4; ++nt) acc2[nt] = z;
    }
    const int jrow = wave * 16 + lr;
    for (int ot = 0; ot < HID / 32; ++ot) {
      const int obase = ot * 32;
      const _Float16* ap = W2h + (size_t)jrow * HID + obase + kbA;
      h8 alo = *(const h8*)ap;
      h8 ahi = *(const h8*)(ap + 16);
      v16h a2 = __builtin_shufflevector(alo, ahi, 0, 1, 2, 3, 4, 5, 6, 7, 8, 9,
                                        10, 11, 12, 13, 14, 15);
      #pragma unroll
      for (int nt = 0; nt < 4; ++nt) {
        const _Float16* bp = &hbuf[(nt * 16 + lr) * 264 + obase + kbB];
        h8 blo = *(const h8*)bp;
        h8 bhi = *(const h8*)(bp + 16);
        v16h b2f = __builtin_shufflevector(blo, bhi, 0, 1, 2, 3, 4, 5, 6, 7, 8,
                                           9, 10, 11, 12, 13, 14, 15);
        acc2[nt] = __builtin_amdgcn_wmma_f32_16x16x32_f16(
            false, a2, false, b2f, (short)0, acc2[nt], false, false);
      }
    }
    // write out[j, pos] position-major with +b2 (rows >= 65 are padding)
    #pragma unroll
    for (int nt = 0; nt < 4; ++nt) {
      const int s = nt * 16 + lr;
      float* orow = outbuf + (size_t)(b * HW + sbase + s) * NOUT;
      #pragma unroll
      for (int e = 0; e < 8; ++e) {
        const int j = wave * 16 + lhi * 8 + e;
        if (j < NOUT) orow[j] = acc2[nt][e] + b2[j];
      }
    }
  }
}

// ---------------------------------------------------------------------------
// Gather + activation epilogue.
// ---------------------------------------------------------------------------
__global__ __launch_bounds__(256) void gather_out(
    const float* __restrict__ outbuf, const int* __restrict__ pos_idx,
    const int* __restrict__ neg_idx, float* __restrict__ conf_out,
    float* __restrict__ offsets, float* __restrict__ class_out) {
  int i = blockIdx.x * blockDim.x + threadIdx.x;
  if (i >= MIDX) return;
  const int AHW = NA * HW;
  {
    int p = pos_idx[i];
    int bb = p / AHW;
    int r = p - bb * AHW;
    int a = r / HW;
    int s = r - a * HW;
    const float* row = outbuf + (size_t)(bb * HW + s) * NOUT;
    conf_out[i] = 1.0f / (1.0f + __expf(-row[a * 5]));
    offsets[4 * i + 0] = 1.0f / (1.0f + __expf(-row[a * 5 + 1])) - 0.5f;
    offsets[4 * i + 1] = 1.0f / (1.0f + __expf(-row[a * 5 + 2])) - 0.5f;
    offsets[4 * i + 2] = row[a * 5 + 3];
    offsets[4 * i + 3] = row[a * 5 + 4];
    #pragma unroll
    for (int c = 0; c < NC; ++c) class_out[NC * i + c] = row[45 + c];
  }
  {
    int p = neg_idx[i];
    int bb = p / AHW;
    int r = p - bb * AHW;
    int a = r / HW;
    int s = r - a * HW;
    conf_out[MIDX + i] =
        1.0f / (1.0f + __expf(-outbuf[(size_t)(bb * HW + s) * NOUT + a * 5]));
  }
}

extern "C" void kernel_launch(void* const* d_in, const int* in_sizes, int n_in,
                              void* d_out, int out_size, void* d_ws,
                              size_t ws_size, hipStream_t stream) {
  (void)in_sizes; (void)n_in; (void)out_size; (void)ws_size;
  const float* features = (const float*)d_in[0];
  const float* W1 = (const float*)d_in[1];
  const float* b1 = (const float*)d_in[2];
  const float* W2 = (const float*)d_in[3];
  const float* b2 = (const float*)d_in[4];
  const int* pos_idx = (const int*)d_in[5];
  const int* neg_idx = (const int*)d_in[6];

  char* ws = (char*)d_ws;
  _Float16* w1h = (_Float16*)(ws + WS_W1H);
  _Float16* w2h = (_Float16*)(ws + WS_W2H);
  float* outbuf = (float*)(ws + WS_OUT);

  prep_weights<<<(HID * CIN + NOUTP * HID + 255) / 256, 256, 0, stream>>>(
      W1, W2, w1h, w2h);
  fused_mlp<<<NB * HW / 64, 256, 0, stream>>>(features, w1h, b1, w2h, b2,
                                              outbuf);

  float* conf_out = (float*)d_out;             // 2*M
  float* offsets = conf_out + 2 * MIDX;        // M*4
  float* class_out = offsets + 4 * MIDX;       // M*20
  gather_out<<<(MIDX + 255) / 256, 256, 0, stream>>>(
      outbuf, pos_idx, neg_idx, conf_out, offsets, class_out);
}
